// SlotAttention_40407052321075
// MI455X (gfx1250) — matrix operand order repaired and checked by reading, not compiled, not run on hardware
//
#include <hip/hip_runtime.h>
#include <cstdint>

#define B_ 64
#define N_ 4096
#define D_ 256
#define S_ 8
#define H_ 128
#define M_ 256
#define SCALE_ 0.08838834764831845f  // 128^-0.5

typedef __attribute__((ext_vector_type(16))) __bf16       v16bf;
typedef __attribute__((ext_vector_type(8)))  float        v8f;
typedef __attribute__((ext_vector_type(4)))  unsigned int u32x4;

union BF16x16 { v16bf v; u32x4 q[2]; };

__device__ __forceinline__ unsigned short f2bf(float f) {
  unsigned int u = __float_as_uint(f);
  u += 0x7FFFu + ((u >> 16) & 1u);   // round-to-nearest-even
  return (unsigned short)(u >> 16);
}

__device__ __forceinline__ float wave_sum(float v) {
#pragma unroll
  for (int off = 16; off > 0; off >>= 1) v += __shfl_xor(v, off, 32);
  return v;
}

// ---------------------------------------------------------------------------
// Kernel 0: convert [Wk;Wv] -> bf16 [256][256] row-major (row = output col),
//           and init slots = mu + (softplus(sigma)+1e-6)*noise
// ---------------------------------------------------------------------------
__global__ __launch_bounds__(256) void prep_kernel(
    const float* __restrict__ Wk, const float* __restrict__ Wv,
    const float* __restrict__ slots_mu, const float* __restrict__ slots_sigma,
    const float* __restrict__ noise, unsigned short* __restrict__ Wkv,
    float* __restrict__ slots) {
  const int idx = blockIdx.x * 256 + threadIdx.x;  // 0..65535
  const int c = idx >> 8, d = idx & 255;
  const float wv = (c < 128) ? Wk[c * 256 + d] : Wv[(c - 128) * 256 + d];
  Wkv[idx] = f2bf(wv);
  const int h = idx & 127;
  const float sg = slots_sigma[h];
  const float sp = (sg > 20.f) ? sg : log1pf(expf(sg));
  slots[idx] = slots_mu[h] + (sp + 1e-6f) * noise[idx];
}

// ---------------------------------------------------------------------------
// Kernel 1: layernorm(inputs) then k = x@Wk^T, v = x@Wv^T via bf16 WMMA.
//   k stored row-major  [B*N][128] bf16  (B operand of logits GEMM)
//   v stored transposed [B][128][4096] bf16 (B operand of updates GEMM)
// One block = 16 token rows, 8 waves, 2 col-tiles each (16 tiles of 16 cols).
// ---------------------------------------------------------------------------
__global__ __launch_bounds__(256) void kv_ln_proj(
    const float* __restrict__ x, const float* __restrict__ g,
    const float* __restrict__ beta, const unsigned short* __restrict__ Wkv,
    unsigned short* __restrict__ kmat, unsigned short* __restrict__ vT) {
  __shared__ __align__(16) float          xf[16][256];
  __shared__ __align__(16) unsigned short xh[16][256];
  const int tid = threadIdx.x;
  const long tok0 = (long)blockIdx.x * 16;
  const int b = (int)(tok0 >> 12);
  const int tloc = (int)(tok0 & 4095);

  for (int i = tid; i < 16 * 256; i += 256)
    xf[i >> 8][i & 255] = x[tok0 * 256 + i];
  __syncthreads();

  const int w = tid >> 5, lane = tid & 31;
#pragma unroll
  for (int rr = 0; rr < 2; ++rr) {  // each wave layernorms 2 rows
    const int row = w * 2 + rr;
    float s1 = 0.f, s2 = 0.f;
#pragma unroll
    for (int k = 0; k < 8; ++k) {
      const float v = xf[row][lane + k * 32];
      s1 += v; s2 += v * v;
    }
    s1 = wave_sum(s1); s2 = wave_sum(s2);
    const float mu = s1 * (1.f / 256.f);
    const float var = s2 * (1.f / 256.f) - mu * mu;
    const float rs = rsqrtf(var + 1e-5f);
#pragma unroll
    for (int k = 0; k < 8; ++k) {
      const int d = lane + k * 32;
      xh[row][d] = f2bf((xf[row][d] - mu) * rs * g[d] + beta[d]);
    }
  }
  __syncthreads();

  const int n = lane & 15, gg = lane >> 4;
  for (int t = w * 2; t < w * 2 + 2; ++t) {
    v8f acc = {};
    const unsigned short* wrow = Wkv + (size_t)(t * 16 + n) * 256;
#pragma unroll
    for (int kk = 0; kk < 8; ++kk) {  // K = 256 in 8 chunks of 32
      BF16x16 A, Bm;
      A.q[0]  = *(const u32x4*)&xh[n][kk * 32 + gg * 8];
      A.q[1]  = *(const u32x4*)&xh[n][kk * 32 + 16 + gg * 8];
      Bm.q[0] = *(const u32x4*)&wrow[kk * 32 + gg * 8];
      Bm.q[1] = *(const u32x4*)&wrow[kk * 32 + 16 + gg * 8];
      acc = __builtin_amdgcn_wmma_f32_16x16x32_bf16(false, A.v, false, Bm.v,
                                                    (short)0, acc, false, false);
    }
    if (t < 8) {  // k output: row-major [tok][h]
#pragma unroll
      for (int r = 0; r < 8; ++r) {
        const long tok = tok0 + gg * 8 + r;
        kmat[tok * 128 + t * 16 + n] = f2bf(acc[r]);
      }
    } else {      // v output: transposed [b][h][tok], 8 halves contiguous
      const int d = (t - 8) * 16 + n;
      union { unsigned short s[8]; u32x4 q; } pk;
#pragma unroll
      for (int r = 0; r < 8; ++r) pk.s[r] = f2bf(acc[r]);
      *(u32x4*)&vT[((size_t)b * 128 + d) * 4096 + tloc + gg * 8] = pk.q;
    }
  }
}

// ---------------------------------------------------------------------------
// Kernel 2: q = layernorm(slots) @ Wq^T  -> bf16 [B][16][128] (rows 8..15 = 0)
//           also zero rowsum[b][s]
// ---------------------------------------------------------------------------
__global__ __launch_bounds__(128) void q_proj(
    const float* __restrict__ slots, const float* __restrict__ g,
    const float* __restrict__ beta, const float* __restrict__ Wq,
    unsigned short* __restrict__ qbuf, float* __restrict__ rowsum) {
  const int b = blockIdx.x >> 4, s = blockIdx.x & 15, tid = threadIdx.x;
  if (tid == 0) rowsum[b * 16 + s] = 0.f;
  if (s >= 8) { qbuf[(size_t)(b * 16 + s) * 128 + tid] = 0; return; }
  __shared__ float sln[128];
  __shared__ float part[8];
  const float v = slots[(size_t)(b * 8 + s) * 128 + tid];
  const float s1 = wave_sum(v), s2 = wave_sum(v * v);
  if ((tid & 31) == 0) { part[tid >> 5] = s1; part[4 + (tid >> 5)] = s2; }
  __syncthreads();
  const float sum = part[0] + part[1] + part[2] + part[3];
  const float sq  = part[4] + part[5] + part[6] + part[7];
  const float mu = sum * (1.f / 128.f);
  const float rs = rsqrtf(sq * (1.f / 128.f) - mu * mu + 1e-5f);
  sln[tid] = (v - mu) * rs * g[tid] + beta[tid];
  __syncthreads();
  float acc = 0.f;
  const float* wr = Wq + (size_t)tid * 128;
  for (int d = 0; d < 128; ++d) acc += sln[d] * wr[d];
  qbuf[(size_t)(b * 16 + s) * 128 + tid] = f2bf(acc);
}

// ---------------------------------------------------------------------------
// Kernel 3: logits = q@k^T * SCALE, softmax over slots (per-lane, 8 accum
// regs), store attn bf16 [B][16][4096], atomic partial rowsums per slot.
// Grid = B * N/128, 8 waves/block, one 16x16 token tile per wave.
// ---------------------------------------------------------------------------
__global__ __launch_bounds__(256) void attn_kernel(
    const unsigned short* __restrict__ qbuf, const unsigned short* __restrict__ kmat,
    unsigned short* __restrict__ attn, float* __restrict__ rowsum) {
  const int b = blockIdx.x >> 5;
  const int w = threadIdx.x >> 5, lane = threadIdx.x & 31;
  const int j0 = (blockIdx.x & 31) * 128 + w * 16;
  const int n = lane & 15, gg = lane >> 4;
  const unsigned short* qrow = qbuf + (size_t)b * 16 * 128 + (size_t)n * 128;
  const unsigned short* krow = kmat + ((size_t)b * 4096 + j0 + n) * 128;
  v8f acc = {};
#pragma unroll
  for (int kk = 0; kk < 4; ++kk) {  // K = H = 128
    BF16x16 A, Bm;
    A.q[0]  = *(const u32x4*)&qrow[kk * 32 + gg * 8];
    A.q[1]  = *(const u32x4*)&qrow[kk * 32 + 16 + gg * 8];
    Bm.q[0] = *(const u32x4*)&krow[kk * 32 + gg * 8];
    Bm.q[1] = *(const u32x4*)&krow[kk * 32 + 16 + gg * 8];
    acc = __builtin_amdgcn_wmma_f32_16x16x32_bf16(false, A.v, false, Bm.v,
                                                  (short)0, acc, false, false);
  }
  // softmax over slots: lanes 0..15 hold rows 0..7 (all 8 slots) per column
  float a[8];
  float mx = -3.4e38f;
#pragma unroll
  for (int r = 0; r < 8; ++r) { a[r] = acc[r] * SCALE_; mx = fmaxf(mx, a[r]); }
  float se = 0.f;
#pragma unroll
  for (int r = 0; r < 8; ++r) { a[r] = __expf(a[r] - mx); se += a[r]; }
  const float inv = (gg == 0) ? (1.f / se) : 0.f;  // pad rows 8..15 -> 0
  const int j = j0 + n;
#pragma unroll
  for (int r = 0; r < 8; ++r) {
    a[r] *= inv;
    attn[((size_t)b * 16 + r + gg * 8) * 4096 + j] = f2bf(a[r]);
  }
#pragma unroll
  for (int r = 0; r < 8; ++r) {  // lanes 16..31 contribute 0
    const float t = wave_sum(a[r]);
    if (lane == 0) atomicAdd(&rowsum[b * 16 + r], t);
  }
}

// ---------------------------------------------------------------------------
// Kernel 4: updates = (attn @ v) / (rowsum + 1e-8). Grid = B, wave w -> 16
// cols of H, K = 4096 in 128 WMMA chunks. Output f32 [B][16][128] rows 0..7.
// ---------------------------------------------------------------------------
__global__ __launch_bounds__(256) void updates_kernel(
    const unsigned short* __restrict__ attn, const unsigned short* __restrict__ vT,
    const float* __restrict__ rowsum, float* __restrict__ updates) {
  const int b = blockIdx.x;
  const int w = threadIdx.x >> 5, lane = threadIdx.x & 31;
  const int n = lane & 15, gg = lane >> 4;
  const unsigned short* arow = attn + (size_t)b * 16 * 4096 + (size_t)n * 4096;
  const unsigned short* vrow = vT + ((size_t)b * 128 + w * 16 + n) * 4096;
  v8f acc = {};
#pragma unroll 4
  for (int kk = 0; kk < 128; ++kk) {
    BF16x16 A, Bm;
    A.q[0]  = *(const u32x4*)&arow[kk * 32 + gg * 8];
    A.q[1]  = *(const u32x4*)&arow[kk * 32 + 16 + gg * 8];
    Bm.q[0] = *(const u32x4*)&vrow[kk * 32 + gg * 8];
    Bm.q[1] = *(const u32x4*)&vrow[kk * 32 + 16 + gg * 8];
    acc = __builtin_amdgcn_wmma_f32_16x16x32_bf16(false, A.v, false, Bm.v,
                                                  (short)0, acc, false, false);
  }
  if (gg == 0) {  // rows 0..7 valid
#pragma unroll
    for (int r = 0; r < 8; ++r)
      updates[((size_t)b * 16 + r) * 128 + w * 16 + n] =
          acc[r] / (rowsum[b * 16 + r] + 1e-8f);
  }
}

// ---------------------------------------------------------------------------
// Kernel 5: GRU cell + LN + MLP(silu) + residual. One block per (b,s).
// ---------------------------------------------------------------------------
__global__ __launch_bounds__(128) void gru_mlp(
    const float* __restrict__ updates, const float* __restrict__ slots_in,
    const float* __restrict__ W_ih, const float* __restrict__ W_hh,
    const float* __restrict__ b_ih, const float* __restrict__ b_hh,
    const float* __restrict__ g, const float* __restrict__ beta,
    const float* __restrict__ W1, const float* __restrict__ b1,
    const float* __restrict__ W2, const float* __restrict__ b2,
    float* __restrict__ out) {
  const int b = blockIdx.x >> 3, s = blockIdx.x & 7, tid = threadIdx.x;
  __shared__ float sx[128], sh[128], sm[256], part[8];
  sx[tid] = updates[(size_t)(b * 16 + s) * 128 + tid];
  const float h = slots_in[(size_t)(b * 8 + s) * 128 + tid];
  sh[tid] = h;
  __syncthreads();
  float gi[3], gh[3];
#pragma unroll
  for (int k = 0; k < 3; ++k) {
    const float* wi = W_ih + (size_t)(k * 128 + tid) * 128;
    const float* wh = W_hh + (size_t)(k * 128 + tid) * 128;
    float ai = b_ih[k * 128 + tid], ah = b_hh[k * 128 + tid];
    for (int d = 0; d < 128; ++d) { ai += sx[d] * wi[d]; ah += sh[d] * wh[d]; }
    gi[k] = ai; gh[k] = ah;
  }
  const float r  = 1.f / (1.f + expf(-(gi[0] + gh[0])));
  const float z  = 1.f / (1.f + expf(-(gi[1] + gh[1])));
  const float nn = tanhf(gi[2] + r * gh[2]);
  const float hnew = (1.f - z) * nn + z * h;
  __syncthreads();  // all reads of sx/sh done before reuse
  const float s1 = wave_sum(hnew), s2 = wave_sum(hnew * hnew);
  if ((tid & 31) == 0) { part[tid >> 5] = s1; part[4 + (tid >> 5)] = s2; }
  __syncthreads();
  const float sum = part[0] + part[1] + part[2] + part[3];
  const float sq  = part[4] + part[5] + part[6] + part[7];
  const float mu = sum * (1.f / 128.f);
  const float rs = rsqrtf(sq * (1.f / 128.f) - mu * mu + 1e-5f);
  sx[tid] = (hnew - mu) * rs * g[tid] + beta[tid];
  __syncthreads();
#pragma unroll
  for (int jj = 0; jj < 2; ++jj) {
    const int m = tid + jj * 128;
    const float* w1r = W1 + (size_t)m * 128;
    float a = b1[m];
    for (int d = 0; d < 128; ++d) a += sx[d] * w1r[d];
    sm[m] = a / (1.f + expf(-a));  // silu
  }
  __syncthreads();
  const float* w2r = W2 + (size_t)tid * 256;
  float o = hnew + b2[tid];
  for (int d = 0; d < 256; ++d) o += sm[d] * w2r[d];
  out[(size_t)(b * 8 + s) * 128 + tid] = o;
}

// ---------------------------------------------------------------------------
extern "C" void kernel_launch(void* const* d_in, const int* in_sizes, int n_in,
                              void* d_out, int out_size, void* d_ws, size_t ws_size,
                              hipStream_t stream) {
  (void)in_sizes; (void)n_in; (void)out_size; (void)ws_size;
  const float* inputs      = (const float*)d_in[0];
  const float* noise       = (const float*)d_in[1];
  const float* slots_mu    = (const float*)d_in[2];
  const float* slots_sigma = (const float*)d_in[3];
  const float* Wq   = (const float*)d_in[4];
  const float* Wk   = (const float*)d_in[5];
  const float* Wv   = (const float*)d_in[6];
  const float* ln_in_g = (const float*)d_in[7];
  const float* ln_in_b = (const float*)d_in[8];
  const float* ln_s_g  = (const float*)d_in[9];
  const float* ln_s_b  = (const float*)d_in[10];
  const float* ln_m_g  = (const float*)d_in[11];
  const float* ln_m_b  = (const float*)d_in[12];
  const float* W_ih = (const float*)d_in[13];
  const float* W_hh = (const float*)d_in[14];
  const float* b_ih = (const float*)d_in[15];
  const float* b_hh = (const float*)d_in[16];
  const float* W1   = (const float*)d_in[17];
  const float* b1   = (const float*)d_in[18];
  const float* W2   = (const float*)d_in[19];
  const float* b2   = (const float*)d_in[20];

  char* p = (char*)d_ws;
  unsigned short* Wkv  = (unsigned short*)p; p += (size_t)256 * 256 * 2;
  unsigned short* kmat = (unsigned short*)p; p += (size_t)B_ * N_ * H_ * 2;
  unsigned short* vT   = (unsigned short*)p; p += (size_t)B_ * N_ * H_ * 2;
  unsigned short* qbuf = (unsigned short*)p; p += (size_t)B_ * 16 * H_ * 2;
  unsigned short* attn = (unsigned short*)p; p += (size_t)B_ * 16 * N_ * 2;
  float* rowsum  = (float*)p; p += (size_t)B_ * 16 * 4;
  float* updates = (float*)p; p += (size_t)B_ * 16 * H_ * 4;
  float* slots   = (float*)p; p += (size_t)B_ * S_ * H_ * 4;

  prep_kernel<<<256, 256, 0, stream>>>(Wk, Wv, slots_mu, slots_sigma, noise,
                                       Wkv, slots);
  kv_ln_proj<<<B_ * N_ / 16, 256, 0, stream>>>(inputs, ln_in_g, ln_in_b, Wkv,
                                               kmat, vT);
  for (int it = 0; it < 3; ++it) {
    q_proj<<<B_ * 16, 128, 0, stream>>>(slots, ln_s_g, ln_s_b, Wq, qbuf, rowsum);
    attn_kernel<<<B_ * (N_ / 128), 256, 0, stream>>>(qbuf, kmat, attn, rowsum);
    updates_kernel<<<B_, 256, 0, stream>>>(attn, vT, rowsum, updates);
    float* dst = (it == 2) ? (float*)d_out : slots;
    gru_mlp<<<B_ * S_, 128, 0, stream>>>(updates, slots, W_ih, W_hh, b_ih, b_hh,
                                         ln_m_g, ln_m_b, W1, b1, W2, b2, dst);
  }
}
